// CREStereo_63359357551215
// MI455X (gfx1250) — matrix operand order repaired and checked by reading, not compiled, not run on hardware
//
#include <hip/hip_runtime.h>
#include <hip/hip_bf16.h>
#include <math.h>

// ---------------------------------------------------------------------------
// Types / helpers
// ---------------------------------------------------------------------------
typedef __attribute__((ext_vector_type(16))) _Float16 v16h;
typedef __attribute__((ext_vector_type(8)))  _Float16 v8h;
typedef __attribute__((ext_vector_type(8)))  float    v8f;
typedef __attribute__((ext_vector_type(4))) unsigned int u32x4;
typedef __attribute__((ext_vector_type(8)))  int i32x8;
typedef __attribute__((ext_vector_type(4)))  int i32x4;

#define TPB 256
static inline int cdiv(long a, long b) { return (int)((a + b - 1) / b); }
static inline long rup64(long k) { return (k + 63) & ~63L; }

enum { ACT_NONE = 0, ACT_RELU = 1, ACT_SIG = 2, ACT_TANH = 3, ACT_SIGOFF = 4, ACT_ELU1 = 5 };

__device__ __forceinline__ float apply_act(float v, int act) {
  switch (act) {
    case ACT_RELU:   return fmaxf(v, 0.f);
    case ACT_SIG:    return 1.f / (1.f + __expf(-v));
    case ACT_TANH:   return tanhf(v);
    case ACT_SIGOFF: return (1.f / (1.f + __expf(-v)) - 0.5f) * 2.f;
    case ACT_ELU1:   return v < 0.f ? __expf(v) : v + 1.f;
  }
  return v;
}

__device__ __forceinline__ float tap0(const float* p, int y, int x, int H, int W) {
  return (y >= 0 && y < H && x >= 0 && x < W) ? p[(size_t)y * W + x] : 0.f;
}
__device__ __forceinline__ float bilin(const float* p, float cy, float cx, int H, int W) {
  float y0f = floorf(cy), x0f = floorf(cx);
  int y0 = (int)y0f, x0 = (int)x0f;
  float wy = cy - y0f, wx = cx - x0f;
  float a = tap0(p, y0, x0, H, W),     b = tap0(p, y0 + 1, x0, H, W);
  float c = tap0(p, y0, x0 + 1, H, W), d = tap0(p, y0 + 1, x0 + 1, H, W);
  return a * (1.f - wx) * (1.f - wy) + b * (1.f - wx) * wy + c * wx * (1.f - wy) + d * wx * wy;
}

// Build a 16-half fragment from two aligned 8-half chunks.
__device__ __forceinline__ v16h frag2(const _Float16* p0, const _Float16* p1) {
  v8h lo = *(const v8h*)p0;
  v8h hi = *(const v8h*)p1;
  return __builtin_shufflevector(lo, hi, 0, 1, 2, 3, 4, 5, 6, 7, 8, 9, 10, 11, 12, 13, 14, 15);
}

// ---------------------------------------------------------------------------
// Implicit-GEMM conv / linear kernel (WMMA f32 <- f16xf16, 16x16x32)
//   A = weights [Cout][Ktp] f16, K padded to 64 and zero-filled -> pure b128 loads
//   B = im2col tile staged in LDS: 32 pixels x 64 K, 16B-aligned rows
//   Workgroup tile: 128 Cout x 32 pixels; each of 8 waves: 16x32 output,
//   two f32 accumulators, 4 x v_wmma per LDS stage.
// ---------------------------------------------------------------------------
__global__ void k_conv_wmma(const _Float16* __restrict__ Wt, const float* __restrict__ bias,
                            const float* __restrict__ X, float* __restrict__ Y,
                            int Nb, int Cin, int Hin, int Win, int Cout, int KH, int KW,
                            int sy, int sx, int py, int px, int OH, int OW, int Ktp, int act) {
  const int KHW  = KH * KW;
  const int Ktot = Cin * KHW;
  const int OHOW = OH * OW;
  const int P    = Nb * OHOW;
  const int pixBase = blockIdx.x * 32;
  const int wave = threadIdx.x >> 5;
  const int lane = threadIdx.x & 31;
  const int half = lane >> 4;
  const int l16  = lane & 15;
  const int coutBase = blockIdx.y * 128 + wave * 16;
  const int row = coutBase + l16;
  const _Float16* arow = Wt + (size_t)(row < Cout ? row : 0) * Ktp;

  __shared__ __align__(16) _Float16 Bt[32][72];   // 32 pixel rows x 64 K (+8 pad)
  __shared__ int pixN[32], pixIY[32], pixIX[32];

  // one-time per-pixel im2col base decomposition
  if (threadIdx.x < 32) {
    int pix = pixBase + threadIdx.x;
    int n = -1, iy0 = 0, ix0 = 0;
    if (pix < P) {
      n = pix / OHOW;
      int rem = pix - n * OHOW;
      int oy = rem / OW, ox = rem - oy * OW;
      iy0 = oy * sy - py;
      ix0 = ox * sx - px;
    }
    pixN[threadIdx.x] = n; pixIY[threadIdx.x] = iy0; pixIX[threadIdx.x] = ix0;
  }
  __syncthreads();

  v8f acc0 = {}, acc1 = {};
  const int kk = threadIdx.x & 63;
  const int colBase = threadIdx.x >> 6;    // 0..3

  for (int k0 = 0; k0 < Ktp; k0 += 64) {
    if (row < Cout && k0 + 64 < Ktp) __builtin_prefetch(arow + k0 + 64, 0, 1);
    // ---- fill B tile: one k-decomposition per thread, 8 pixel gathers ----
    {
      int k = k0 + kk;
      bool kvalid = k < Ktot;
      int cin = 0, kh = 0, kw = 0;
      if (kvalid) {
        if (KHW == 1) { cin = k; }
        else { cin = k / KHW; int rr = k - cin * KHW; kh = rr / KW; kw = rr - kh * KW; }
      }
#pragma unroll
      for (int cc = 0; cc < 8; ++cc) {
        int col = colBase + 4 * cc;
        float v = 0.f;
        if (kvalid) {
          int n = pixN[col];
          if (n >= 0) {
            int iy = pixIY[col] + kh, ix = pixIX[col] + kw;
            if (iy >= 0 && iy < Hin && ix >= 0 && ix < Win)
              v = X[(((size_t)n * Cin + cin) * Hin + iy) * Win + ix];
          }
        }
        Bt[col][kk] = (_Float16)v;
      }
    }
    __syncthreads();
    // ---- 4 MMAs: 2 K-steps x 2 pixel tiles ----
    v16h a0 = frag2(arow + k0 +      half * 16, arow + k0 +      half * 16 + 8);
    v16h a1 = frag2(arow + k0 + 32 + half * 16, arow + k0 + 32 + half * 16 + 8);
    v16h b00 = frag2(&Bt[l16][half * 16],           &Bt[l16][half * 16 + 8]);
    v16h b01 = frag2(&Bt[l16][32 + half * 16],      &Bt[l16][32 + half * 16 + 8]);
    v16h b10 = frag2(&Bt[16 + l16][half * 16],      &Bt[16 + l16][half * 16 + 8]);
    v16h b11 = frag2(&Bt[16 + l16][32 + half * 16], &Bt[16 + l16][32 + half * 16 + 8]);
    acc0 = __builtin_amdgcn_wmma_f32_16x16x32_f16(false, a0, false, b00, (short)0, acc0, false, false);
    acc1 = __builtin_amdgcn_wmma_f32_16x16x32_f16(false, a0, false, b10, (short)0, acc1, false, false);
    acc0 = __builtin_amdgcn_wmma_f32_16x16x32_f16(false, a1, false, b01, (short)0, acc0, false, false);
    acc1 = __builtin_amdgcn_wmma_f32_16x16x32_f16(false, a1, false, b11, (short)0, acc1, false, false);
    __syncthreads();
  }

  // ---- store both 16-pixel tiles (fused bias + activation) ----
#pragma unroll
  for (int t = 0; t < 2; ++t) {
    int pix = pixBase + t * 16 + l16;
    if (pix < P) {
      int n = pix / OHOW, rem = pix - n * OHOW;
      int oy = rem / OW, ox = rem - oy * OW;
      const v8f& acc = t ? acc1 : acc0;
#pragma unroll
      for (int vg = 0; vg < 8; ++vg) {
        int m = coutBase + vg + 8 * half;
        if (m < Cout) {
          float v = acc[vg] + (bias ? bias[m] : 0.f);
          Y[(((size_t)n * Cout + m) * OH + oy) * OW + ox] = apply_act(v, act);
        }
      }
    }
  }
}

// ---------------------------------------------------------------------------
// Weight pack: f32 [Cout][Ktot] -> f16 [Cout][Ktp] zero-padded
// ---------------------------------------------------------------------------
__global__ void k_pack_w(const float* __restrict__ src, _Float16* __restrict__ dst,
                         int Ktot, int Ktp, long n) {
  long i = (long)blockIdx.x * TPB + threadIdx.x;
  if (i >= n) return;
  long r = i / Ktp;
  int k = (int)(i - r * Ktp);
  dst[i] = (k < Ktot) ? (_Float16)src[r * Ktot + k] : (_Float16)0.f;
}

// ---------------------------------------------------------------------------
// TDM / cluster probe kernel (compiled, not launched): gfx1250 async-tensor path.
// ---------------------------------------------------------------------------
#if defined(__has_builtin)
#if __has_builtin(__builtin_amdgcn_tensor_load_to_lds)
#define HAVE_TDM 1
#endif
#endif

__global__ void k_tdm_probe(const float* __restrict__ src, float* __restrict__ dst) {
  __shared__ float lds[TPB];
#ifdef HAVE_TDM
  u32x4 g0 = {};
  i32x8 g1 = {};
  i32x4 g2 = {}, g3 = {};
#if __clang_major__ >= 23
  i32x8 g4 = {};
  __builtin_amdgcn_tensor_load_to_lds(g0, g1, g2, g3, g4, 0);
#else
  __builtin_amdgcn_tensor_load_to_lds(g0, g1, g2, g3, 0);
#endif
  __builtin_amdgcn_s_wait_tensorcnt(0);
#endif
  __builtin_amdgcn_s_cluster_barrier();
  lds[threadIdx.x] = src[threadIdx.x];
  __syncthreads();
  dst[threadIdx.x] = lds[threadIdx.x ^ 1];
}

// ---------------------------------------------------------------------------
// Elementwise / reduction kernels
// ---------------------------------------------------------------------------
__global__ void k_zero(float* p, long n) {
  long i = (long)blockIdx.x * TPB + threadIdx.x;
  if (i < n) p[i] = 0.f;
}
__global__ void k_prep(const float* im1, const float* im2, float* x, long nper) {
  long i = (long)blockIdx.x * TPB + threadIdx.x;
  if (i < nper)            x[i] = 2.f * (im1[i] / 255.f) - 1.f;
  else if (i < 2 * nper)   x[i] = 2.f * (im2[i - nper] / 255.f) - 1.f;
}
__global__ void k_act(const float* x, float* y, long n, int act) {
  long i = (long)blockIdx.x * TPB + threadIdx.x;
  if (i < n) y[i] = apply_act(x[i], act);
}
__global__ void k_add(const float* a, const float* b, float* y, long n, int relu) {
  long i = (long)blockIdx.x * TPB + threadIdx.x;
  if (i < n) { float v = a[i] + b[i]; y[i] = relu ? fmaxf(v, 0.f) : v; }
}
__global__ void k_mul_sig(const float* r, const float* net, float* y, long n) {
  long i = (long)blockIdx.x * TPB + threadIdx.x;
  if (i < n) y[i] = net[i] / (1.f + __expf(-r[i]));
}
__global__ void k_gru(float* net, const float* z, const float* q, long n) {
  long i = (long)blockIdx.x * TPB + threadIdx.x;
  if (i < n) {
    float zz = 1.f / (1.f + __expf(-z[i]));
    net[i] = (1.f - zz) * net[i] + zz * tanhf(q[i]);
  }
}
__global__ void k_inorm(float* x, int HW, int relu) {
  float* p = x + (size_t)blockIdx.x * HW;
  __shared__ float s1[TPB], s2[TPB];
  float a = 0.f, a2 = 0.f;
  for (int i = threadIdx.x; i < HW; i += TPB) { float v = p[i]; a += v; a2 += v * v; }
  s1[threadIdx.x] = a; s2[threadIdx.x] = a2; __syncthreads();
  for (int o = TPB / 2; o > 0; o >>= 1) {
    if ((int)threadIdx.x < o) { s1[threadIdx.x] += s1[threadIdx.x + o]; s2[threadIdx.x] += s2[threadIdx.x + o]; }
    __syncthreads();
  }
  float m = s1[0] / HW;
  float inv = rsqrtf(s2[0] / HW - m * m + 1e-5f);
  for (int i = threadIdx.x; i < HW; i += TPB) {
    float v = (p[i] - m) * inv;
    p[i] = relu ? fmaxf(v, 0.f) : v;
  }
}
__global__ void k_lnorm(const float* x, const float* g, const float* b, float* y, int D, int L) {
  int l = blockIdx.x;
  __shared__ float s1[TPB], s2[TPB];
  float a = 0.f, a2 = 0.f;
  for (int c = threadIdx.x; c < D; c += TPB) { float v = x[(size_t)c * L + l]; a += v; a2 += v * v; }
  s1[threadIdx.x] = a; s2[threadIdx.x] = a2; __syncthreads();
  for (int o = TPB / 2; o > 0; o >>= 1) {
    if ((int)threadIdx.x < o) { s1[threadIdx.x] += s1[threadIdx.x + o]; s2[threadIdx.x] += s2[threadIdx.x + o]; }
    __syncthreads();
  }
  float m = s1[0] / D;
  float inv = rsqrtf(s2[0] / D - m * m + 1e-5f);
  for (int c = threadIdx.x; c < D; c += TPB)
    y[(size_t)c * L + l] = (x[(size_t)c * L + l] - m) * inv * g[c] + b[c];
}
__global__ void k_avgpool(const float* x, float* y, int C, int H, int W, int k) {
  int OH = H / k, OW = W / k;
  long total = (long)C * OH * OW;
  long idx = (long)blockIdx.x * TPB + threadIdx.x;
  if (idx >= total) return;
  int c = (int)(idx / (OH * OW)); int r = (int)(idx % (OH * OW));
  int oy = r / OW, ox = r % OW;
  float s = 0.f;
  for (int i = 0; i < k; ++i)
    for (int j = 0; j < k; ++j)
      s += x[((size_t)c * H + oy * k + i) * W + ox * k + j];
  y[idx] = s / (float)(k * k);
}
__global__ void k_pe(float* pe, int H, int W) {
  long n = 256L * H * W;
  long idx = (long)blockIdx.x * TPB + threadIdx.x;
  if (idx >= n) return;
  int c = (int)(idx / (H * W)); int p = (int)(idx % (H * W));
  int y = p / W, x = p % W;
  int i = c >> 2, m = c & 3;
  float div = __expf((float)(2 * i) * (-logf(10000.f) / 128.f));
  float xv = (float)(x + 1) * div, yv = (float)(y + 1) * div;
  float v = (m == 0) ? __sinf(xv) : (m == 1) ? __cosf(xv) : (m == 2) ? __sinf(yv) : __cosf(yv);
  pe[idx] = v;
}
__global__ void k_rowsum(const float* x, float* out, int L) {
  int r = blockIdx.x;
  __shared__ float sb[TPB];
  float a = 0.f;
  for (int i = threadIdx.x; i < L; i += TPB) a += x[(size_t)r * L + i];
  sb[threadIdx.x] = a; __syncthreads();
  for (int o = TPB / 2; o > 0; o >>= 1) {
    if ((int)threadIdx.x < o) sb[threadIdx.x] += sb[threadIdx.x + o];
    __syncthreads();
  }
  if (threadIdx.x == 0) out[r] = sb[0];
}
// KV[h][d][v] = sum_s K[h*32+d][s] * V[h*32+v][s] / L      (K,V are [256][L])
__global__ void k_kv(const float* K, const float* V, float* KV, int L) {
  int idx = blockIdx.x * TPB + threadIdx.x;
  if (idx >= 8 * 32 * 32) return;
  int h = idx >> 10, d = (idx >> 5) & 31, v = idx & 31;
  const float* kr = K + (size_t)(h * 32 + d) * L;
  const float* vr = V + (size_t)(h * 32 + v) * L;
  float s = 0.f;
  for (int i = 0; i < L; ++i) s += kr[i] * vr[i];
  KV[idx] = s / (float)L;
}
// msg[h*32+v][l] = L * Z(l,h) * sum_d Q[h*32+d][l] * KV[h][d][v]
__global__ void k_msg(const float* Q, const float* KV, const float* Ksum, float* msg, int L) {
  long idx = (long)blockIdx.x * TPB + threadIdx.x;
  if (idx >= 256L * L) return;
  int l = (int)(idx % L); int ch = (int)(idx / L);
  int h = ch >> 5, v = ch & 31;
  float zden = 1e-6f, acc = 0.f;
  for (int d = 0; d < 32; ++d) {
    float q = Q[(size_t)(h * 32 + d) * L + l];
    zden += q * Ksum[h * 32 + d];
    acc  += q * KV[((h * 32 + d) << 5) + v];
  }
  msg[(size_t)ch * L + l] = (float)L * acc / zden;
}
__global__ void k_warp(const float* f2, const float* flow, float* out, int C, int H, int W) {
  long n = (long)C * H * W;
  long idx = (long)blockIdx.x * TPB + threadIdx.x;
  if (idx >= n) return;
  int HW = H * W;
  int c = (int)(idx / HW); int p = (int)(idx % HW);
  int y = p / W, x = p % W;
  float cx = (float)x + flow[p];
  float cy = (float)y + flow[HW + p];
  out[idx] = bilin(f2 + (size_t)c * HW, cy, cx, H, W);
}
// corr_iter correlation: f1 . shifted(edge-clamped) f2w, 4 groups x 9 offsets
__global__ void k_get_corr(const float* f1, const float* f2w, float* out, int H, int W, int small) {
  int HW = H * W;
  long n = 36L * HW;
  long idx = (long)blockIdx.x * TPB + threadIdx.x;
  if (idx >= n) return;
  int g = (int)(idx / (9 * HW)); int r = (int)(idx % (9 * HW));
  int o = r / HW; int p = r % HW;
  int y = p / W, x = p % W;
  int py = small ? 1 : 0, px = small ? 1 : 4;
  int dy = (o / (2 * px + 1)) - py;
  int dx = (o % (2 * px + 1)) - px;
  int yy = min(max(y + dy, 0), H - 1);
  int xx = min(max(x + dx, 0), W - 1);
  float s = 0.f;
  for (int c = 0; c < 64; ++c) {
    size_t ch = (size_t)(g * 64 + c) * HW;
    s += f1[ch + p] * f2w[ch + (size_t)yy * W + xx];
  }
  out[(size_t)(g * 9 + o) * HW + p] = s * (1.f / 64.f);
}
// corr_att_offset correlation: bilinear-sampled f2 at base+off+extra_offset
__global__ void k_corr_offset(const float* f1, const float* f2, const float* flow,
                              const float* eo, float* out, int H, int W, int small) {
  int HW = H * W;
  long n = 36L * HW;
  long idx = (long)blockIdx.x * TPB + threadIdx.x;
  if (idx >= n) return;
  int g = (int)(idx / (9 * HW)); int r = (int)(idx % (9 * HW));
  int o = r / HW; int p = r % HW;
  int y = p / W, x = p % W;
  float dx, dy;
  if (small) { dx = (float)(o % 3 - 1); dy = (float)(o / 3 - 1); }
  else       { dx = (float)(o - 4);     dy = 0.f; }
  float cx = (float)x + flow[p]      + dx + eo[(size_t)(o * 2 + 0) * HW + p];
  float cy = (float)y + flow[HW + p] + dy + eo[(size_t)(o * 2 + 1) * HW + p];
  float y0f = floorf(cy), x0f = floorf(cx);
  int y0 = (int)y0f, x0 = (int)x0f;
  float wy = cy - y0f, wx = cx - x0f;
  float wa = (1.f - wx) * (1.f - wy), wb = (1.f - wx) * wy, wc = wx * (1.f - wy), wd = wx * wy;
  float s = 0.f;
  for (int c = 0; c < 64; ++c) {
    const float* f2c = f2 + (size_t)(g * 64 + c) * HW;
    float sv = tap0(f2c, y0, x0, H, W) * wa + tap0(f2c, y0 + 1, x0, H, W) * wb +
               tap0(f2c, y0, x0 + 1, H, W) * wc + tap0(f2c, y0 + 1, x0 + 1, H, W) * wd;
    s += f1[(size_t)(g * 64 + c) * HW + p] * sv;
  }
  out[(size_t)(g * 9 + o) * HW + p] = s * (1.f / 64.f);
}
// convex upsample (rate 4), with mask prescale (0.25) folded in and output scale
__global__ void k_convex(const float* flow, const float* mask, float* out,
                         int H, int W, float mscale, float oscale) {
  int OH = 4 * H, OW = 4 * W;
  long n = 2L * OH * OW;
  long idx = (long)blockIdx.x * TPB + threadIdx.x;
  if (idx >= n) return;
  int c = (int)(idx / (OH * OW)); int r = (int)(idx % (OH * OW));
  int y = r / OW, x = r % OW;
  int h = y >> 2, i = y & 3, w = x >> 2, j = x & 3;
  int HW = H * W;
  float mv[9], mx = -1e30f;
#pragma unroll
  for (int k = 0; k < 9; ++k) {
    mv[k] = mscale * mask[(size_t)(k * 16 + i * 4 + j) * HW + h * W + w];
    mx = fmaxf(mx, mv[k]);
  }
  float den = 0.f;
#pragma unroll
  for (int k = 0; k < 9; ++k) { mv[k] = __expf(mv[k] - mx); den += mv[k]; }
  float s = 0.f;
#pragma unroll
  for (int k = 0; k < 9; ++k) {
    int hh = h + k / 3 - 1, ww = w + k % 3 - 1;
    float fv = (hh >= 0 && hh < H && ww >= 0 && ww < W) ? 4.f * flow[(size_t)c * HW + hh * W + ww] : 0.f;
    s += mv[k] * fv;
  }
  out[idx] = oscale * s / den;
}
__global__ void k_interp(const float* x, float* y, int C, int H, int W, int OH, int OW, float scale) {
  long n = (long)C * OH * OW;
  long idx = (long)blockIdx.x * TPB + threadIdx.x;
  if (idx >= n) return;
  int c = (int)(idx / (OH * OW)); int r = (int)(idx % (OH * OW));
  int oy = r / OW, ox = r % OW;
  float fy = (OH > 1) ? (float)(H - 1) * oy / (float)(OH - 1) : 0.f;
  float fx = (OW > 1) ? (float)(W - 1) * ox / (float)(OW - 1) : 0.f;
  y[idx] = scale * bilin(x + (size_t)c * H * W, fy, fx, H, W);
}

// ---------------------------------------------------------------------------
// Host-side orchestration helpers
// ---------------------------------------------------------------------------
struct Arena {
  char* p; size_t off;
  float* f(size_t n) { float* r = (float*)(p + off); off = (off + n * 4 + 255) & ~(size_t)255; return r; }
  _Float16* h(size_t n) { _Float16* r = (_Float16*)(p + off); off = (off + n * 2 + 255) & ~(size_t)255; return r; }
};

static void conv(hipStream_t s, const _Float16* w, const float* b, const float* x, float* y,
                 int Nb, int Cin, int Hin, int Win, int Cout, int KH, int KW,
                 int sy, int sx, int py, int px, int act) {
  int OH = (Hin + 2 * py - KH) / sy + 1;
  int OW = (Win + 2 * px - KW) / sx + 1;
  long P = (long)Nb * OH * OW;
  int Ktp = (int)rup64((long)Cin * KH * KW);
  dim3 g(cdiv(P, 32), cdiv(Cout, 128));
  k_conv_wmma<<<g, TPB, 0, s>>>(w, b, x, y, Nb, Cin, Hin, Win, Cout, KH, KW, sy, sx, py, px, OH, OW, Ktp, act);
}
static void gemm(hipStream_t s, const _Float16* w, const float* b, const float* x, float* y,
                 int Cout, int Cin, int L, int act) {
  conv(s, w, b, x, y, 1, Cin, 1, L, Cout, 1, 1, 1, 1, 0, 0, act);
}
static void inorm(hipStream_t s, float* x, int NC, int HW, int relu) {
  k_inorm<<<NC, TPB, 0, s>>>(x, HW, relu);
}
static void d2d(hipStream_t s, float* dst, const float* src, long n) {
  hipMemcpyAsync(dst, src, (size_t)n * 4, hipMemcpyDeviceToDevice, s);
}

struct RB { const _Float16 *c1, *c2, *d; const float *c1b, *c2b, *db; bool down; };
struct Att { const _Float16 *q, *k, *v, *merge, *mlp1, *mlp2;
             const float *ln1g, *ln1b, *ln2g, *ln2b; };
struct Upd {
  const _Float16 *cc1, *cc2, *cf1, *cf2, *cm, *z1, *r1, *q1, *z2, *r2, *q2, *fh1, *fh2, *mk1, *mk2;
  const float *cc1b, *cc2b, *cf1b, *cf2b, *cmb, *z1b, *r1b, *q1b, *z2b, *r2b, *q2b, *fh1b, *fh2b, *mk1b, *mk2b;
};
struct LScr { float *q, *k, *v, *ksum, *kv, *msg, *tmp, *cat, *h512; };
struct UScr { float *cor, *cor2, *flo, *flo2, *cat, *mf, *x, *hx, *z, *r, *rn, *q; };

static void resblock(hipStream_t s, const RB& r, float* xbuf, float* t1, float* t2,
                     int Nb, int Cin, int& H, int& W, int Cout, int stride) {
  int OH = (H + 2 - 3) / stride + 1;
  int OW = (W + 2 - 3) / stride + 1;
  long n = (long)Nb * Cout * OH * OW;
  conv(s, r.c1, r.c1b, xbuf, t1, Nb, Cin, H, W, Cout, 3, 3, stride, stride, 1, 1, ACT_NONE);
  inorm(s, t1, Nb * Cout, OH * OW, 1);
  conv(s, r.c2, r.c2b, t1, t2, Nb, Cout, OH, OW, Cout, 3, 3, 1, 1, 1, 1, ACT_NONE);
  inorm(s, t2, Nb * Cout, OH * OW, 1);
  if (r.down) {
    conv(s, r.d, r.db, xbuf, t1, Nb, Cin, H, W, Cout, 1, 1, stride, stride, 0, 0, ACT_NONE);
    inorm(s, t1, Nb * Cout, OH * OW, 0);
    k_add<<<cdiv(n, TPB), TPB, 0, s>>>(t1, t2, xbuf, n, 1);
  } else {
    k_add<<<cdiv(n, TPB), TPB, 0, s>>>(xbuf, t2, xbuf, n, 1);
  }
  H = OH; W = OW;
}

static void loftr(hipStream_t s, const Att& w, const float* x, const float* src,
                  float* out, int L, LScr& sc) {
  gemm(s, w.q, nullptr, x,   sc.q, 256, 256, L, ACT_NONE);
  gemm(s, w.k, nullptr, src, sc.k, 256, 256, L, ACT_NONE);
  gemm(s, w.v, nullptr, src, sc.v, 256, 256, L, ACT_NONE);
  long nQ = 256L * L;
  k_act<<<cdiv(nQ, TPB), TPB, 0, s>>>(sc.q, sc.q, nQ, ACT_ELU1);
  k_act<<<cdiv(nQ, TPB), TPB, 0, s>>>(sc.k, sc.k, nQ, ACT_ELU1);
  k_rowsum<<<256, TPB, 0, s>>>(sc.k, sc.ksum, L);
  k_kv<<<cdiv(8192, TPB), TPB, 0, s>>>(sc.k, sc.v, sc.kv, L);
  k_msg<<<cdiv(nQ, TPB), TPB, 0, s>>>(sc.q, sc.kv, sc.ksum, sc.msg, L);
  gemm(s, w.merge, nullptr, sc.msg, sc.tmp, 256, 256, L, ACT_NONE);
  k_lnorm<<<L, TPB, 0, s>>>(sc.tmp, w.ln1g, w.ln1b, sc.tmp, 256, L);
  d2d(s, sc.cat, x, nQ);
  d2d(s, sc.cat + nQ, sc.tmp, nQ);
  gemm(s, w.mlp1, nullptr, sc.cat, sc.h512, 512, 512, L, ACT_RELU);
  gemm(s, w.mlp2, nullptr, sc.h512, sc.tmp, 256, 512, L, ACT_NONE);
  k_lnorm<<<L, TPB, 0, s>>>(sc.tmp, w.ln2g, w.ln2b, sc.tmp, 256, L);
  k_add<<<cdiv(nQ, TPB), TPB, 0, s>>>(x, sc.tmp, out, nQ, 0);
}

static void update_block(hipStream_t s, const Upd& u, float* net, const float* inp,
                         const float* corr, const float* flow, int H, int W,
                         float* mask_out, float* dflow_out, UScr& sc) {
  long HW = (long)H * W;
  conv(s, u.cc1, u.cc1b, corr, sc.cor, 1, 36, H, W, 256, 1, 1, 1, 1, 0, 0, ACT_RELU);
  conv(s, u.cc2, u.cc2b, sc.cor, sc.cor2, 1, 256, H, W, 192, 3, 3, 1, 1, 1, 1, ACT_RELU);
  conv(s, u.cf1, u.cf1b, flow, sc.flo, 1, 2, H, W, 128, 7, 7, 1, 1, 3, 3, ACT_RELU);
  conv(s, u.cf2, u.cf2b, sc.flo, sc.flo2, 1, 128, H, W, 64, 3, 3, 1, 1, 1, 1, ACT_RELU);
  d2d(s, sc.cat, sc.cor2, 192 * HW);
  d2d(s, sc.cat + 192 * HW, sc.flo2, 64 * HW);
  conv(s, u.cm, u.cmb, sc.cat, sc.mf, 1, 256, H, W, 126, 3, 3, 1, 1, 1, 1, ACT_RELU);
  d2d(s, sc.x, inp, 128 * HW);
  d2d(s, sc.x + 128 * HW, sc.mf, 126 * HW);
  d2d(s, sc.x + 254 * HW, flow, 2 * HW);
  for (int pass = 0; pass < 2; ++pass) {
    int kh = pass ? 5 : 1, kw = pass ? 1 : 5;
    int py = pass ? 2 : 0, px = pass ? 0 : 2;
    const _Float16 *wz = pass ? u.z2 : u.z1, *wr = pass ? u.r2 : u.r1, *wq = pass ? u.q2 : u.q1;
    const float *bz = pass ? u.z2b : u.z1b, *br = pass ? u.r2b : u.r1b, *bq = pass ? u.q2b : u.q1b;
    d2d(s, sc.hx, net, 128 * HW);
    d2d(s, sc.hx + 128 * HW, sc.x, 256 * HW);
    conv(s, wz, bz, sc.hx, sc.z, 1, 384, H, W, 128, kh, kw, 1, 1, py, px, ACT_NONE);
    conv(s, wr, br, sc.hx, sc.r, 1, 384, H, W, 128, kh, kw, 1, 1, py, px, ACT_NONE);
    k_mul_sig<<<cdiv(128 * HW, TPB), TPB, 0, s>>>(sc.r, net, sc.rn, 128 * HW);
    d2d(s, sc.hx, sc.rn, 128 * HW);
    d2d(s, sc.hx + 128 * HW, sc.x, 256 * HW);
    conv(s, wq, bq, sc.hx, sc.q, 1, 384, H, W, 128, kh, kw, 1, 1, py, px, ACT_NONE);
    k_gru<<<cdiv(128 * HW, TPB), TPB, 0, s>>>(net, sc.z, sc.q, 128 * HW);
  }
  conv(s, u.fh1, u.fh1b, net, sc.cor, 1, 128, H, W, 256, 3, 3, 1, 1, 1, 1, ACT_RELU);
  conv(s, u.fh2, u.fh2b, sc.cor, dflow_out, 1, 256, H, W, 2, 3, 3, 1, 1, 1, 1, ACT_NONE);
  conv(s, u.mk1, u.mk1b, net, sc.cor, 1, 128, H, W, 256, 3, 3, 1, 1, 1, 1, ACT_RELU);
  conv(s, u.mk2, u.mk2b, sc.cor, mask_out, 1, 256, H, W, 144, 1, 1, 1, 1, 0, 0, ACT_NONE);
}

// ---------------------------------------------------------------------------
// kernel_launch: full CREStereo forward (N=1, 384x512, iters=6 -> 12 preds)
// ---------------------------------------------------------------------------
extern "C" void kernel_launch(void* const* d_in, const int* in_sizes, int n_in,
                              void* d_out, int out_size, void* d_ws, size_t ws_size,
                              hipStream_t stream) {
  (void)in_sizes; (void)n_in; (void)out_size;
  Arena A{(char*)d_ws, 0};
  hipStream_t s = stream;
  int ii = 0;
  const float* image1 = (const float*)d_in[ii++];
  const float* image2 = (const float*)d_in[ii++];
  auto nextf = [&]() { return (const float*)d_in[ii++]; };
  // pack one weight tensor: f32 [cout][ktot] -> f16 [cout][rup64(ktot)]
  auto PW = [&](long cout, long ktot) -> const _Float16* {
    const float* src = nextf();
    long ktp = rup64(ktot);
    long n = cout * ktp;
    _Float16* d = A.h(n);
    k_pack_w<<<cdiv(n, TPB), TPB, 0, s>>>(src, d, (int)ktot, (int)ktp, n);
    return d;
  };

  // ---- unpack params (setup_inputs dict order) ----
  const _Float16* conv1w = PW(64, 3 * 49); const float* conv1b = nextf();
  RB rbs[6];
  const int rcin[6]  = {64, 64, 64, 96, 96, 128};
  const int rcout[6] = {64, 64, 96, 96, 128, 128};
  const bool rdn[6]  = {false, false, true, false, true, false};
  for (int i = 0; i < 6; ++i) {
    rbs[i].c1 = PW(rcout[i], (long)rcin[i] * 9);  rbs[i].c1b = nextf();
    rbs[i].c2 = PW(rcout[i], (long)rcout[i] * 9); rbs[i].c2b = nextf();
    rbs[i].down = rdn[i]; rbs[i].d = nullptr; rbs[i].db = nullptr;
    if (rdn[i]) { rbs[i].d = PW(rcout[i], rcin[i]); rbs[i].db = nextf(); }
  }
  const _Float16* conv2w = PW(256, 128); const float* conv2b = nextf();
  auto loadAtt = [&]() {
    Att a;
    a.q = PW(256, 256); a.k = PW(256, 256); a.v = PW(256, 256); a.merge = PW(256, 256);
    a.mlp1 = PW(512, 512); a.mlp2 = PW(256, 512);
    a.ln1g = nextf(); a.ln1b = nextf(); a.ln2g = nextf(); a.ln2b = nextf();
    return a;
  };
  Att satt = loadAtt();
  Att catt = loadAtt();
  Upd u;
  u.cc1 = PW(256, 36);       u.cc1b = nextf();
  u.cc2 = PW(192, 256 * 9);  u.cc2b = nextf();
  u.cf1 = PW(128, 2 * 49);   u.cf1b = nextf();
  u.cf2 = PW(64, 128 * 9);   u.cf2b = nextf();
  u.cm  = PW(126, 256 * 9);  u.cmb  = nextf();
  u.z1 = PW(128, 384 * 5); u.z1b = nextf();
  u.r1 = PW(128, 384 * 5); u.r1b = nextf();
  u.q1 = PW(128, 384 * 5); u.q1b = nextf();
  u.z2 = PW(128, 384 * 5); u.z2b = nextf();
  u.r2 = PW(128, 384 * 5); u.r2b = nextf();
  u.q2 = PW(128, 384 * 5); u.q2b = nextf();
  u.fh1 = PW(256, 128 * 9); u.fh1b = nextf();
  u.fh2 = PW(2, 256 * 9);   u.fh2b = nextf();
  u.mk1 = PW(256, 128 * 9); u.mk1b = nextf();
  u.mk2 = PW(144, 256);     u.mk2b = nextf();
  const _Float16* off8w  = PW(18, 256 * 9); const float* off8b  = nextf();
  const _Float16* off16w = PW(18, 256 * 9); const float* off16b = nextf();
  // iters input is on-device; the pipeline below is hardcoded for iters=6.

  // ---- encoder (batch 2: [im1; im2]) ----
  const int HF = 384, WF = 512;
  const long HWF = (long)HF * WF;
  float* x0  = A.f(2 * 3 * HWF);
  float* big0 = A.f(2 * 64 * HWF);
  float* big1 = A.f(2 * 64 * HWF);
  float* big2 = A.f(2 * 64 * HWF);
  k_prep<<<cdiv(2 * 3 * HWF, TPB), TPB, 0, s>>>(image1, image2, x0, 3 * HWF);
  conv(s, conv1w, conv1b, x0, big0, 2, 3, HF, WF, 64, 7, 7, 1, 1, 3, 3, ACT_NONE);
  inorm(s, big0, 2 * 64, HF * WF, 1);
  int H = HF, W = WF;
  resblock(s, rbs[0], big0, big1, big2, 2, 64, H, W, 64, 1);
  resblock(s, rbs[1], big0, big1, big2, 2, 64, H, W, 64, 1);
  resblock(s, rbs[2], big0, big1, big2, 2, 64, H, W, 96, 2);
  resblock(s, rbs[3], big0, big1, big2, 2, 96, H, W, 96, 1);
  resblock(s, rbs[4], big0, big1, big2, 2, 96, H, W, 128, 2);
  resblock(s, rbs[5], big0, big1, big2, 2, 128, H, W, 128, 1);
  const int H4 = H, W4 = W;                       // 96 x 128
  const long HW4 = (long)H4 * W4;
  float* fmap = A.f(2 * 256 * HW4);
  conv(s, conv2w, conv2b, big0, fmap, 2, 128, H4, W4, 256, 1, 1, 1, 1, 0, 0, ACT_NONE);
  float* fmap1 = fmap;
  float* fmap2 = fmap + 256 * HW4;

  // ---- pools, context, offsets ----
  const int H8 = H4 / 2, W8 = W4 / 2;  const long HW8 = (long)H8 * W8;
  const int H16 = H4 / 4, W16v = W4 / 4; const long HW16 = (long)H16 * W16v;
  const int L16 = (int)HW16;           // 768 tokens
  float* f1d8 = A.f(256 * HW8);  float* f2d8 = A.f(256 * HW8);
  float* f116 = A.f(256 * HW16); float* f216 = A.f(256 * HW16);
  k_avgpool<<<cdiv(256 * HW8, TPB), TPB, 0, s>>>(fmap1, f1d8, 256, H4, W4, 2);
  k_avgpool<<<cdiv(256 * HW8, TPB), TPB, 0, s>>>(fmap2, f2d8, 256, H4, W4, 2);
  k_avgpool<<<cdiv(256 * HW16, TPB), TPB, 0, s>>>(fmap1, f116, 256, H4, W4, 4);
  k_avgpool<<<cdiv(256 * HW16, TPB), TPB, 0, s>>>(fmap2, f216, 256, H4, W4, 4);
  float* off8  = A.f(18 * HW8);
  float* off16 = A.f(18 * HW16);
  conv(s, off8w, off8b, f1d8, off8, 1, 256, H8, W8, 18, 3, 3, 1, 1, 1, 1, ACT_SIGOFF);
  conv(s, off16w, off16b, f116, off16, 1, 256, H16, W16v, 18, 3, 3, 1, 1, 1, 1, ACT_SIGOFF);
  float* net = A.f(128 * HW4);  float* inp = A.f(128 * HW4);
  k_act<<<cdiv(128 * HW4, TPB), TPB, 0, s>>>(fmap1, net, 128 * HW4, ACT_TANH);
  k_act<<<cdiv(128 * HW4, TPB), TPB, 0, s>>>(fmap1 + 128 * HW4, inp, 128 * HW4, ACT_RELU);
  float* net8  = A.f(128 * HW8);  float* inp8  = A.f(128 * HW8);
  float* net16 = A.f(128 * HW16); float* inp16 = A.f(128 * HW16);
  k_avgpool<<<cdiv(128 * HW8, TPB), TPB, 0, s>>>(net, net8, 128, H4, W4, 2);
  k_avgpool<<<cdiv(128 * HW8, TPB), TPB, 0, s>>>(inp, inp8, 128, H4, W4, 2);
  k_avgpool<<<cdiv(128 * HW16, TPB), TPB, 0, s>>>(net, net16, 128, H4, W4, 4);
  k_avgpool<<<cdiv(128 * HW16, TPB), TPB, 0, s>>>(inp, inp16, 128, H4, W4, 4);

  // ---- positional encoding + self attention at 1/16 ----
  float* pe = A.f(256 * HW16);
  k_pe<<<cdiv(256 * HW16, TPB), TPB, 0, s>>>(pe, H16, W16v);
  float* ain = A.f(256 * HW16); float* bin = A.f(256 * HW16);
  k_add<<<cdiv(256 * HW16, TPB), TPB, 0, s>>>(f116, pe, ain, 256 * HW16, 0);
  k_add<<<cdiv(256 * HW16, TPB), TPB, 0, s>>>(f216, pe, bin, 256 * HW16, 0);
  LScr ls;
  ls.q = A.f(256L * L16); ls.k = A.f(256L * L16); ls.v = A.f(256L * L16);
  ls.ksum = A.f(256); ls.kv = A.f(8192); ls.msg = A.f(256L * L16);
  ls.tmp = A.f(256L * L16); ls.cat = A.f(512L * L16); ls.h512 = A.f(512L * L16);
  float* f1a = A.f(256 * HW16); float* f2a = A.f(256 * HW16);
  loftr(s, satt, ain, ain, f1a, L16, ls);
  loftr(s, satt, bin, bin, f2a, L16, ls);

  // ---- shared iteration scratch (sized for the largest stage: 96x128) ----
  UScr us;
  us.cor = A.f(256 * HW4); us.cor2 = A.f(192 * HW4); us.flo = A.f(128 * HW4);
  us.flo2 = A.f(64 * HW4); us.cat = A.f(256 * HW4);  us.mf = A.f(126 * HW4);
  us.x = A.f(256 * HW4);   us.hx = A.f(384 * HW4);   us.z = A.f(128 * HW4);
  us.r = A.f(128 * HW4);   us.rn = A.f(128 * HW4);   us.q = A.f(128 * HW4);
  float* corr  = A.f(36 * HW4);
  float* maskb = A.f(144 * HW4);
  float* dflow = A.f(2 * HW4);
  float* up    = A.f(2L * HF * WF);
  float* a2 = A.f(256 * HW16); float* b2 = A.f(256 * HW16);
  float* outp = (float*)d_out;
  const long SZO = 2L * HF * WF;
  int pred = 0;

  // ---- stage 1: 1/16, cross attention + offset correlation (3 iters) ----
  float* flow16 = A.f(2 * HW16);
  k_zero<<<cdiv(2 * HW16, TPB), TPB, 0, s>>>(flow16, 2 * HW16);
  for (int it = 0; it < 3; ++it) {
    int sp = (it % 2 == 1);
    loftr(s, catt, f1a, f2a, a2, L16, ls);
    loftr(s, catt, f2a, f1a, b2, L16, ls);
    k_corr_offset<<<cdiv(36 * HW16, TPB), TPB, 0, s>>>(a2, b2, flow16, off16, corr, H16, W16v, sp);
    update_block(s, u, net16, inp16, corr, flow16, H16, W16v, maskb, dflow, us);
    k_add<<<cdiv(2 * HW16, TPB), TPB, 0, s>>>(flow16, dflow, flow16, 2 * HW16, 0);
    k_convex<<<cdiv(2L * 16 * HW16, TPB), TPB, 0, s>>>(flow16, maskb, up, H16, W16v, 0.25f, 1.f);
    k_interp<<<cdiv(SZO, TPB), TPB, 0, s>>>(up, outp + (pred++) * SZO, 2, 4 * H16, 4 * W16v, HF, WF, -4.f);
  }
  // hand off: flow8 = -(H8 / (4*H16)) * interp(up, H8, W8)
  float* flow8 = A.f(2 * HW8);
  k_interp<<<cdiv(2 * HW8, TPB), TPB, 0, s>>>(up, flow8, 2, 4 * H16, 4 * W16v, H8, W8,
                                              -((float)H8 / (float)(4 * H16)));

  // ---- stage 2: 1/8, offset correlation without attention (3 iters) ----
  for (int it = 0; it < 3; ++it) {
    int sp = (it % 2 == 1);
    k_corr_offset<<<cdiv(36 * HW8, TPB), TPB, 0, s>>>(f1d8, f2d8, flow8, off8, corr, H8, W8, sp);
    update_block(s, u, net8, inp8, corr, flow8, H8, W8, maskb, dflow, us);
    k_add<<<cdiv(2 * HW8, TPB), TPB, 0, s>>>(flow8, dflow, flow8, 2 * HW8, 0);
    k_convex<<<cdiv(2L * 16 * HW8, TPB), TPB, 0, s>>>(flow8, maskb, up, H8, W8, 0.25f, 1.f);
    k_interp<<<cdiv(SZO, TPB), TPB, 0, s>>>(up, outp + (pred++) * SZO, 2, 4 * H8, 4 * W8, HF, WF, -2.f);
  }
  float* flow4 = A.f(2 * HW4);
  k_interp<<<cdiv(2 * HW4, TPB), TPB, 0, s>>>(up, flow4, 2, 4 * H8, 4 * W8, H4, W4,
                                              -((float)H4 / (float)(4 * H8)));

  // ---- stage 3: 1/4, warped window correlation (6 iters) ----
  float* f2w = A.f(256 * HW4);
  for (int it = 0; it < 6; ++it) {
    int sp = (it % 2 == 1);
    k_warp<<<cdiv(256 * HW4, TPB), TPB, 0, s>>>(fmap2, flow4, f2w, 256, H4, W4);
    k_get_corr<<<cdiv(36 * HW4, TPB), TPB, 0, s>>>(fmap1, f2w, corr, H4, W4, sp);
    update_block(s, u, net, inp, corr, flow4, H4, W4, maskb, dflow, us);
    k_add<<<cdiv(2 * HW4, TPB), TPB, 0, s>>>(flow4, dflow, flow4, 2 * HW4, 0);
    k_convex<<<cdiv(SZO, TPB), TPB, 0, s>>>(flow4, maskb, outp + (pred++) * SZO, H4, W4, 0.25f, -1.f);
  }
  (void)ws_size;
}
